// GQA_28518582845792
// MI455X (gfx1250) — matrix operand order repaired and checked by reading, not compiled, not run on hardware
//
#include <hip/hip_runtime.h>

#define B_  2
#define T_  2048
#define DM  2048
#define QH  16
#define KVH 4
#define HD  128
#define NG  4
#define BT  (B_ * T_)

typedef __attribute__((ext_vector_type(16))) __bf16 v16bf;
typedef __attribute__((ext_vector_type(8)))  float  v8f;
typedef __attribute__((ext_vector_type(4)))  int    i32x4;

union FragU { uint4 q[2]; v16bf v; };
union AccU  { v8f v; float f[8]; };

typedef __attribute__((address_space(1))) i32x4* gptr128;
typedef __attribute__((address_space(3))) i32x4* lptr128;

#if __has_builtin(__builtin_amdgcn_global_load_async_to_lds_b128)
#define HAS_ASYNC_LDS 1
#else
#define HAS_ASYNC_LDS 0
#endif

// 16B global -> LDS copy: async DMA path on CDNA5, sync fallback otherwise.
__device__ inline void g2l_b128(const unsigned short* g, unsigned short* l) {
#if HAS_ASYNC_LDS
  __builtin_amdgcn_global_load_async_to_lds_b128((gptr128)g, (lptr128)l, 0, 0);
#else
  *(uint4*)l = *(const uint4*)g;
#endif
}

__device__ inline void wait_async_zero() {
#if __has_builtin(__builtin_amdgcn_s_wait_asynccnt)
  __builtin_amdgcn_s_wait_asynccnt(0);
#elif HAS_ASYNC_LDS
  asm volatile("s_wait_asynccnt 0x0" ::: "memory");
#endif
}

__device__ inline v8f v8zero() {
  v8f z = {0.f, 0.f, 0.f, 0.f, 0.f, 0.f, 0.f, 0.f};
  return z;
}

__device__ inline v8f wmma_bf16(v16bf a, v16bf b, v8f c) {
  // (neg_a, A, neg_b, B, c_mod, C, reuse_a, reuse_b)
  return __builtin_amdgcn_wmma_f32_16x16x32_bf16(false, a, false, b, (short)0, c,
                                                 false, false);
}

__device__ inline unsigned short f2bf(float f) {
  unsigned u = __float_as_uint(f);
  u += 0x7fffu + ((u >> 16) & 1u);   // round-to-nearest-even
  return (unsigned short)(u >> 16);
}

// ---------------------------------------------------------------------------
// f32 -> bf16 elementwise convert
// ---------------------------------------------------------------------------
__global__ void cvt_bf16_kernel(const float* __restrict__ in,
                                unsigned short* __restrict__ out, int n) {
  int i = blockIdx.x * blockDim.x + threadIdx.x;
  if (i < n) out[i] = f2bf(in[i]);
}

// ---------------------------------------------------------------------------
// Generic bf16 GEMM: C[M,N] (f32) = A[M,K] * B[N,K]^T, all row-major.
// Block: 64x64 tile, 128 threads = 4 waves, each wave a 32x32 subtile
// (2x2 WMMA tiles). BK = 64 (2 K-steps of 32 per staged tile).
// Double-buffered LDS fed by global_load_async_to_lds_b128; the async DMA
// for tile kt+1 overlaps the WMMA work on tile kt (ASYNCcnt pipelining).
// ---------------------------------------------------------------------------
__global__ __launch_bounds__(128)
void gemm_bf16_kernel(const unsigned short* __restrict__ A,
                      const unsigned short* __restrict__ Bm,
                      float* __restrict__ C, int M, int N, int K) {
  __shared__ unsigned short As[2][64][72];  // row stride 144B (16B aligned)
  __shared__ unsigned short Bs[2][64][72];

  const int tid  = threadIdx.x;
  const int wave = tid >> 5, lane = tid & 31;
  const int wm   = wave >> 1, wn = wave & 1;
  const int hi   = lane >> 4, ln = lane & 15;
  const int mBase = blockIdx.y * 64, nBase = blockIdx.x * 64;

  // Per-thread cooperative-load coordinates: 64x64 bf16 tile = 512 16B chunks;
  // each of 128 threads moves 4 chunks per matrix (rows r0+16i, fixed column).
  const int r0 = tid >> 3;              // 0..15
  const int c0 = (tid & 7) << 3;        // 0,8,...,56
  const unsigned short* gA = A + (size_t)(mBase + r0) * K + c0;
  const unsigned short* gB = Bm + (size_t)(nBase + r0) * K + c0;
  const size_t rstep = (size_t)16 * K;  // 16 rows

  AccU acc[2][2];
  for (int i = 0; i < 2; ++i)
    for (int j = 0; j < 2; ++j) acc[i][j].v = v8zero();

  auto issue = [&](int kt, int buf) {
    const unsigned short* a = gA + kt * 64;
    const unsigned short* b = gB + kt * 64;
#pragma unroll
    for (int i = 0; i < 4; ++i) {
      g2l_b128(a + i * rstep, &As[buf][r0 + 16 * i][c0]);
      g2l_b128(b + i * rstep, &Bs[buf][r0 + 16 * i][c0]);
    }
  };

  const int ksteps = K >> 6;
  issue(0, 0);
  for (int kt = 0; kt < ksteps; ++kt) {
    const int cur = kt & 1;
    wait_async_zero();   // tile kt resident in LDS
    __syncthreads();     // all waves past previous compute; writes visible
    if (kt + 1 < ksteps) issue(kt + 1, cur ^ 1);  // overlap DMA with WMMA

    const unsigned short(*Ac)[72] = As[cur];
    const unsigned short(*Bc)[72] = Bs[cur];
#pragma unroll
    for (int ks = 0; ks < 2; ++ks) {
      FragU af[2], bf[2];
#pragma unroll
      for (int mt = 0; mt < 2; ++mt) {  // A-frag: M = lane%16, K half-wave split
        const unsigned short* p = &Ac[wm * 32 + mt * 16 + ln][ks * 32];
        af[mt].q[0] = *(const uint4*)&p[hi * 8];
        af[mt].q[1] = *(const uint4*)&p[16 + hi * 8];
      }
#pragma unroll
      for (int nt = 0; nt < 2; ++nt) {  // B-frag: N = lane%16, K half-wave split
        const unsigned short* p = &Bc[wn * 32 + nt * 16 + ln][ks * 32];
        bf[nt].q[0] = *(const uint4*)&p[hi * 8];
        bf[nt].q[1] = *(const uint4*)&p[16 + hi * 8];
      }
#pragma unroll
      for (int mt = 0; mt < 2; ++mt)
#pragma unroll
        for (int nt = 0; nt < 2; ++nt)
          acc[mt][nt].v = wmma_bf16(af[mt].v, bf[nt].v, acc[mt][nt].v);
    }
    // no trailing barrier: next iteration's wait+barrier fences buffer reuse
  }

  // C/D layout: VGPR r -> M = r + (lane<16 ? 0 : 8), N = lane%16
  for (int mt = 0; mt < 2; ++mt)
    for (int nt = 0; nt < 2; ++nt)
      for (int r = 0; r < 8; ++r) {
        int row = mBase + wm * 32 + mt * 16 + hi * 8 + r;
        int col = nBase + wn * 32 + nt * 16 + ln;
        C[(size_t)row * N + col] = acc[mt][nt].f[r];
      }
}

// ---------------------------------------------------------------------------
// RoPE: in [B, T, heads*HD] f32 -> out [B, heads, T, HD] bf16 with rotation.
// ---------------------------------------------------------------------------
__global__ void rope_kernel(const float* __restrict__ in,
                            unsigned short* __restrict__ out, int heads) {
  int idx   = blockIdx.x * blockDim.x + threadIdx.x;
  int total = B_ * T_ * heads * (HD / 2);
  if (idx >= total) return;
  int i = idx & 63;        // pair index within head dim
  int r = idx >> 6;
  int h = r % heads;  r /= heads;
  int t = r % T_;
  int b = r / T_;
  const float* p = in + (size_t)(b * T_ + t) * (heads * HD) + h * HD + 2 * i;
  float xe = p[0], xo = p[1];
  float inv = __powf(10000.0f, -(float)i * (1.0f / 64.0f));
  float ang = (float)t * inv;
  float s, c;
  __sincosf(ang, &s, &c);
  size_t ob = ((size_t)(b * heads + h) * T_ + t) * HD + 2 * i;
  out[ob]     = f2bf(xe * c - xo * s);
  out[ob + 1] = f2bf(xe * s + xo * c);
}

// ---------------------------------------------------------------------------
// V transpose: in [B, T, KVH*HD] f32 -> out [B, KVH, HD, T] bf16
// ---------------------------------------------------------------------------
__global__ void vtrans_kernel(const float* __restrict__ in,
                              unsigned short* __restrict__ out) {
  int idx = blockIdx.x * blockDim.x + threadIdx.x;
  if (idx >= B_ * T_ * KVH * HD) return;
  int d  = idx & 127;
  int kv = (idx >> 7) & 3;
  int t  = (idx >> 9) % T_;
  int b  = (idx >> 9) / T_;
  float v = in[(size_t)(b * T_ + t) * (KVH * HD) + kv * HD + d];
  out[((size_t)(b * KVH + kv) * HD + d) * T_ + t] = f2bf(v);
}

// ---------------------------------------------------------------------------
// Flash-style causal GQA attention.
// Grid: (T/64 query tiles, QH, B). Block: 128 threads = 4 waves; each wave
// owns 16 query rows. Per 64-key tile: S = Q K^T (16 WMMAs), online softmax
// (shfl_xor row reductions), P restaged via per-wave LDS, O += P V (16 WMMAs).
// K and V^T tiles staged with async LDS DMA.
// ---------------------------------------------------------------------------
__global__ __launch_bounds__(128)
void attn_kernel(const unsigned short* __restrict__ Qb,
                 const unsigned short* __restrict__ Kb,
                 const unsigned short* __restrict__ Vtb,
                 unsigned short* __restrict__ Ob) {
  __shared__ unsigned short Ks[64][136];    // keys x d (row stride 272B)
  __shared__ unsigned short Vts[128][72];   // d x keys (row stride 144B)
  __shared__ unsigned short Ps[4][16][72];  // per-wave P staging

  const int tid = threadIdx.x, w = tid >> 5, lane = tid & 31;
  const int hi = lane >> 4, ln = lane & 15;
  const int qt = blockIdx.x, h = blockIdx.y, b = blockIdx.z;
  const int kv = h / NG;  // repeat_interleave: q-head h uses kv-head h/NG

  const unsigned short* Qh = Qb + (size_t)(b * QH + h) * T_ * HD;
  const unsigned short* Kh = Kb + (size_t)(b * KVH + kv) * T_ * HD;
  const unsigned short* Vh = Vtb + (size_t)(b * KVH + kv) * HD * T_;

  // Per-thread cooperative-load coordinates.
  const int kr0 = tid >> 4, kc0 = (tid & 15) << 3;  // K tile: rows kr0+8i
  const int vr0 = tid >> 3, vc0 = (tid & 7) << 3;   // V^T tile: rows vr0+16i
  const unsigned short* gK0 = Kh + (size_t)kr0 * HD + kc0;
  const unsigned short* gV0 = Vh + (size_t)vr0 * T_ + vc0;

  // Preload this wave's Q fragments (16 rows x 128d = 4 K-steps), kept live.
  const int qrow = qt * 64 + w * 16;
  FragU qf[4];
#pragma unroll
  for (int ks = 0; ks < 4; ++ks) {
    const unsigned short* p = &Qh[(size_t)(qrow + ln) * HD + ks * 32];
    qf[ks].q[0] = *(const uint4*)&p[hi * 8];
    qf[ks].q[1] = *(const uint4*)&p[16 + hi * 8];
  }

  float m[8], l[8];
  AccU o[8];
  for (int r = 0; r < 8; ++r) { m[r] = -3.0e38f; l[r] = 0.f; }
  for (int d = 0; d < 8; ++d) o[d].v = v8zero();

  const float scale = 0.08838834764831845f;  // 128^-0.5

  for (int kb = 0; kb <= qt; ++kb) {  // causal: only tiles at/below diagonal
    {
      const unsigned short* kg = gK0 + (size_t)kb * 64 * HD;
      const unsigned short* vg = gV0 + kb * 64;
#pragma unroll
      for (int i = 0; i < 8; ++i) {  // 64x128 K tile
        g2l_b128(kg + (size_t)(8 * i) * HD, &Ks[kr0 + 8 * i][kc0]);
      }
#pragma unroll
      for (int i = 0; i < 8; ++i) {  // 128x64 V^T tile
        g2l_b128(vg + (size_t)(16 * i) * T_, &Vts[vr0 + 16 * i][vc0]);
      }
    }
    wait_async_zero();
    __syncthreads();

    // S = Q K^T : 4 N-tiles of 16 keys, contraction over d in 4 steps of 32.
    AccU s[4];
#pragma unroll
    for (int j = 0; j < 4; ++j) {
      s[j].v = v8zero();
#pragma unroll
      for (int ks = 0; ks < 4; ++ks) {
        FragU bf;
        const unsigned short* p = &Ks[j * 16 + ln][ks * 32];
        bf.q[0] = *(const uint4*)&p[hi * 8];
        bf.q[1] = *(const uint4*)&p[16 + hi * 8];
        s[j].v = wmma_bf16(qf[ks].v, bf.v, s[j].v);
      }
    }

    // Scale + causal mask + online softmax. Row M lives in 16 lanes of one
    // half-wave, so xor-shuffles 1/2/4/8 reduce within the row.
    const int qg0 = qt * 64 + w * 16 + hi * 8;
    const int kg0 = kb * 64 + ln;
    float mnew[8], alpha[8];
    for (int r = 0; r < 8; ++r) {
      float mt = -3.0e38f;
#pragma unroll
      for (int j = 0; j < 4; ++j) {
        float v = s[j].f[r] * scale;
        if (kg0 + j * 16 > qg0 + r) v = -3.0e38f;
        s[j].f[r] = v;
        mt = fmaxf(mt, v);
      }
      mt = fmaxf(mt, __shfl_xor(mt, 1, 32));
      mt = fmaxf(mt, __shfl_xor(mt, 2, 32));
      mt = fmaxf(mt, __shfl_xor(mt, 4, 32));
      mt = fmaxf(mt, __shfl_xor(mt, 8, 32));
      mnew[r]  = fmaxf(m[r], mt);
      alpha[r] = __expf(m[r] - mnew[r]);
      m[r]     = mnew[r];
    }
    for (int r = 0; r < 8; ++r) {
      float rs = 0.f;
#pragma unroll
      for (int j = 0; j < 4; ++j) {
        float p = __expf(s[j].f[r] - mnew[r]);
        rs += p;
        Ps[w][hi * 8 + r][j * 16 + ln] = f2bf(p);  // C-layout -> LDS
      }
      rs += __shfl_xor(rs, 1, 32);
      rs += __shfl_xor(rs, 2, 32);
      rs += __shfl_xor(rs, 4, 32);
      rs += __shfl_xor(rs, 8, 32);
      l[r] = l[r] * alpha[r] + rs;
    }
    for (int d = 0; d < 8; ++d)
      for (int r = 0; r < 8; ++r) o[d].f[r] *= alpha[r];

    // O += P V : re-read P as A-fragments (per-wave region, in-order LDS).
#pragma unroll
    for (int ks2 = 0; ks2 < 2; ++ks2) {
      FragU af;
      const unsigned short* pp = &Ps[w][ln][ks2 * 32];
      af.q[0] = *(const uint4*)&pp[hi * 8];
      af.q[1] = *(const uint4*)&pp[16 + hi * 8];
#pragma unroll
      for (int d = 0; d < 8; ++d) {
        FragU bf;
        const unsigned short* vp = &Vts[d * 16 + ln][ks2 * 32];
        bf.q[0] = *(const uint4*)&vp[hi * 8];
        bf.q[1] = *(const uint4*)&vp[16 + hi * 8];
        o[d].v = wmma_bf16(af.v, bf.v, o[d].v);
      }
    }
    __syncthreads();  // tile buffers reused next iteration
  }

  // Normalize and emit attention output as bf16 [B*T, DM] (col = h*128 + d),
  // which is exactly the A-matrix layout for the Wo projection GEMM.
  for (int d = 0; d < 8; ++d)
    for (int r = 0; r < 8; ++r) {
      int trow = qt * 64 + w * 16 + hi * 8 + r;
      int col  = h * HD + d * 16 + ln;
      Ob[(size_t)(b * T_ + trow) * DM + col] = f2bf(o[d].f[r] / l[r]);
    }
}

// ---------------------------------------------------------------------------
// Host-side orchestration
// ---------------------------------------------------------------------------
extern "C" void kernel_launch(void* const* d_in, const int* in_sizes, int n_in,
                              void* d_out, int out_size, void* d_ws,
                              size_t ws_size, hipStream_t stream) {
  const float* x  = (const float*)d_in[0];
  const float* Wq = (const float*)d_in[1];
  const float* Wk = (const float*)d_in[2];
  const float* Wv = (const float*)d_in[3];
  const float* Wo = (const float*)d_in[4];

  char* ws   = (char*)d_ws;
  size_t off = 0;
  auto alloc = [&](size_t bytes) -> void* {
    void* p = ws + off;
    off = (off + bytes + 255) & ~(size_t)255;
    return p;
  };

  unsigned short* xb   = (unsigned short*)alloc((size_t)BT * DM * 2);
  unsigned short* wqb  = (unsigned short*)alloc((size_t)DM * DM * 2);
  unsigned short* wkb  = (unsigned short*)alloc((size_t)KVH * HD * DM * 2);
  unsigned short* wvb  = (unsigned short*)alloc((size_t)KVH * HD * DM * 2);
  unsigned short* wob  = (unsigned short*)alloc((size_t)DM * DM * 2);
  float*          Qf   = (float*)alloc((size_t)BT * DM * 4);
  float*          Kf   = (float*)alloc((size_t)BT * KVH * HD * 4);
  float*          Vf   = (float*)alloc((size_t)BT * KVH * HD * 4);
  unsigned short* Qbf  = (unsigned short*)alloc((size_t)BT * DM * 2);
  unsigned short* Kbf  = (unsigned short*)alloc((size_t)BT * KVH * HD * 2);
  unsigned short* Vtbf = (unsigned short*)alloc((size_t)BT * KVH * HD * 2);
  unsigned short* Obf  = (unsigned short*)Qf;  // alias: Qf dead after RoPE

  auto cvt = [&](const float* in, unsigned short* out, size_t n) {
    cvt_bf16_kernel<<<(unsigned)((n + 255) / 256), 256, 0, stream>>>(in, out, (int)n);
  };
  cvt(x, xb, (size_t)BT * DM);
  cvt(Wq, wqb, (size_t)DM * DM);
  cvt(Wk, wkb, (size_t)KVH * HD * DM);
  cvt(Wv, wvb, (size_t)KVH * HD * DM);
  cvt(Wo, wob, (size_t)DM * DM);

  // QKV projections (C = A * W^T).
  gemm_bf16_kernel<<<dim3(DM / 64, BT / 64), 128, 0, stream>>>(xb, wqb, Qf, BT, DM, DM);
  gemm_bf16_kernel<<<dim3(KVH * HD / 64, BT / 64), 128, 0, stream>>>(xb, wkb, Kf, BT, KVH * HD, DM);
  gemm_bf16_kernel<<<dim3(KVH * HD / 64, BT / 64), 128, 0, stream>>>(xb, wvb, Vf, BT, KVH * HD, DM);

  // RoPE + layout changes.
  {
    size_t nq = (size_t)B_ * T_ * QH * (HD / 2);
    rope_kernel<<<(unsigned)((nq + 255) / 256), 256, 0, stream>>>(Qf, Qbf, QH);
    size_t nk = (size_t)B_ * T_ * KVH * (HD / 2);
    rope_kernel<<<(unsigned)((nk + 255) / 256), 256, 0, stream>>>(Kf, Kbf, KVH);
    size_t nv = (size_t)B_ * T_ * KVH * HD;
    vtrans_kernel<<<(unsigned)((nv + 255) / 256), 256, 0, stream>>>(Vf, Vtbf);
  }

  // Causal GQA attention.
  attn_kernel<<<dim3(T_ / 64, QH, B_), 128, 0, stream>>>(Qbf, Kbf, Vtbf, Obf);

  // Output projection into f32 d_out.
  gemm_bf16_kernel<<<dim3(DM / 64, BT / 64), 128, 0, stream>>>(Obf, wob, (float*)d_out, BT, DM, DM);
}
